// MessagePassingNet_36275293782879
// MI455X (gfx1250) — matrix operand order repaired and checked by reading, not compiled, not run on hardware
//
#include <hip/hip_runtime.h>
#include <hip/hip_bf16.h>

// ---------------------------------------------------------------------------
// CDNA5 (gfx1250) message-passing net.
// GEMMs on v_wmma_f32_16x16x32_bf16. States gathered as bf16 (16B b128 loads),
// accumulated with f32 hardware atomics, re-converted to a bf16 mirror once
// per step (streaming) instead of paying f32 width on every gather touch.
// ---------------------------------------------------------------------------

typedef __attribute__((ext_vector_type(16))) __bf16 v16bf;
typedef __attribute__((ext_vector_type(8)))  __bf16 v8bf;
typedef __attribute__((ext_vector_type(8)))  float  v8f;

#define D_DIM 128
#define H_DIM 64

static __device__ inline v8f wmma_bf16(v16bf a, v16bf b, v8f c) {
    // (neg_a, A, neg_b, B, c_mod, C, reuse_a, reuse_b)
    return __builtin_amdgcn_wmma_f32_16x16x32_bf16(false, a, false, b, (short)0, c,
                                                   false, false);
}

// A fragment (16x32 bf16) for layer 1 from the bf16 state mirror.
// lane holds M = lane&15; K groups are 8-contiguous (16B) and never cross the
// dst/src boundary at k=128.
static __device__ inline v16bf load_a_edge(const __bf16* __restrict__ s,
                                           int dstRow, int srcRow, int kc, int lane) {
    const int kbase = (lane >> 4) * 8;
    const int g0 = kc * 32 + kbase;
    const int g1 = kc * 32 + 16 + kbase;
    const __bf16* p0 = (g0 < D_DIM) ? (s + (size_t)dstRow * D_DIM + g0)
                                    : (s + (size_t)srcRow * D_DIM + (g0 - D_DIM));
    const __bf16* p1 = (g1 < D_DIM) ? (s + (size_t)dstRow * D_DIM + g1)
                                    : (s + (size_t)srcRow * D_DIM + (g1 - D_DIM));
    const v8bf lo = *(const v8bf*)p0;   // 16B, global_load_b128
    const v8bf hi = *(const v8bf*)p1;   // 16B, global_load_b128
    v16bf a;
#pragma unroll
    for (int i = 0; i < 8; ++i) a[i] = lo[i];
#pragma unroll
    for (int i = 0; i < 8; ++i) a[8 + i] = hi[i];
    return a;
}

// A fragment from a 16 x 64 bf16 LDS tile (row-major), K-chunk kc (0 or 1).
static __device__ inline v16bf load_a_lds(const __bf16* __restrict__ Mrows,
                                          int kc, int lane) {
    const int m = lane & 15;
    const int kbase = (lane >> 4) * 8;
    const __bf16* r = Mrows + m * H_DIM;
    v16bf a;
#pragma unroll
    for (int i = 0; i < 8; ++i) a[i] = r[kc * 32 + kbase + i];
#pragma unroll
    for (int i = 0; i < 8; ++i) a[8 + i] = r[kc * 32 + 16 + kbase + i];
    return a;
}

// B fragment (32x16 bf16): lane holds col N = nt*16 + (lane&15), K in table order (8h+s, 16+8h+s-8).
static __device__ inline v16bf load_b_lds(const __bf16* __restrict__ W, int ncols,
                                          int kc, int nt, int lane) {
    const __bf16* p = W + (kc * 32 + (lane >> 4) * 8) * ncols + nt * 16 + (lane & 15);
    v16bf b;
#pragma unroll
    for (int i = 0; i < 16; ++i) b[i] = p[((i & 7) + (i >> 3) * 16) * ncols];
    return b;
}

// B fragment straight from global f32 weights (small head layers).
static __device__ inline v16bf load_b_gmem(const float* __restrict__ W, int ncols,
                                           int kc, int nt, int lane) {
    const float* p = W + (size_t)(kc * 32 + (lane >> 4) * 8) * ncols + nt * 16 + (lane & 15);
    v16bf b;
#pragma unroll
    for (int i = 0; i < 16; ++i) b[i] = (__bf16)p[(size_t)((i & 7) + (i >> 3) * 16) * ncols];
    return b;
}

// ---------------------------------------------------------------------------
// f32 -> bf16 streaming conversion (4 elements / thread).
// ---------------------------------------------------------------------------
__global__ __launch_bounds__(256) void cvt_bf16_kernel(
    const float* __restrict__ in, __bf16* __restrict__ out, int n4)
{
    const int idx = blockIdx.x * blockDim.x + threadIdx.x;
    if (idx >= n4) return;
    const float4 f = ((const float4*)in)[idx];
    v8bf* o = (v8bf*)out;  // unused type; store as 4 packed bf16 (8B)
    __bf16 tmp[4] = { (__bf16)f.x, (__bf16)f.y, (__bf16)f.z, (__bf16)f.w };
    ((uint2*)o)[idx] = *(const uint2*)tmp;
}

// ---------------------------------------------------------------------------
// Per-edge 3-layer MLP + scatter-add. 128 threads = 4 waves; each wave owns a
// 16-edge tile; block covers 64 edges. Weights converted f32->bf16 into LDS.
// ---------------------------------------------------------------------------
__global__ __launch_bounds__(128) void edge_mlp_kernel(
    const __bf16* __restrict__ s_in, float* __restrict__ s_out,
    const float* __restrict__ Win,  const float* __restrict__ b_in,
    const float* __restrict__ Wh,   const float* __restrict__ b_h,
    const float* __restrict__ Wout, const float* __restrict__ b_out,
    const int* __restrict__ src, const int* __restrict__ dst, int E)
{
    __shared__ __bf16 ldsWin[2 * D_DIM * H_DIM];   // 256x64  (32 KB)
    __shared__ __bf16 ldsWh[H_DIM * H_DIM];        // 64x64   ( 8 KB)
    __shared__ __bf16 ldsWout[H_DIM * D_DIM];      // 64x128  (16 KB)
    __shared__ __bf16 ldsM[4][16 * H_DIM];         // per-wave relayout ( 8 KB)

    const int tid = threadIdx.x;
    for (int i = tid; i < 2 * D_DIM * H_DIM; i += 128) ldsWin[i]  = (__bf16)Win[i];
    for (int i = tid; i < H_DIM * H_DIM;     i += 128) ldsWh[i]   = (__bf16)Wh[i];
    for (int i = tid; i < H_DIM * D_DIM;     i += 128) ldsWout[i] = (__bf16)Wout[i];
    __syncthreads();

    const int wave = tid >> 5;
    const int lane = tid & 31;
    const int mrow = lane & 15;
    const int base = blockIdx.x * 64 + wave * 16;

    const int eg = base + mrow;
    const int ec = (eg < E) ? eg : (E - 1);
    const int dRow = dst[ec];
    const int sRow = src[ec];

    __bf16* Mw = ldsM[wave];

    // ---- layer 1: [16,256] @ [256,64] -------------------------------------
    v8f acc1[4];
#pragma unroll
    for (int nt = 0; nt < 4; ++nt) acc1[nt] = (v8f)0.0f;
#pragma unroll
    for (int kc = 0; kc < 8; ++kc) {
        v16bf a = load_a_edge(s_in, dRow, sRow, kc, lane);
#pragma unroll
        for (int nt = 0; nt < 4; ++nt) {
            v16bf b = load_b_lds(ldsWin, H_DIM, kc, nt, lane);
            acc1[nt] = wmma_bf16(a, b, acc1[nt]);
        }
    }
#pragma unroll
    for (int nt = 0; nt < 4; ++nt) {
        const float bias = b_in[nt * 16 + (lane & 15)];
#pragma unroll
        for (int r = 0; r < 8; ++r) {
            float v = fmaxf(acc1[nt][r] + bias, 0.0f);
            const int row = r + 8 * (lane >> 4);
            Mw[row * H_DIM + nt * 16 + (lane & 15)] = (__bf16)v;
        }
    }
    __syncthreads();

    // ---- layer 2: [16,64] @ [64,64] ---------------------------------------
    v8f acc2[4];
#pragma unroll
    for (int nt = 0; nt < 4; ++nt) acc2[nt] = (v8f)0.0f;
#pragma unroll
    for (int kc = 0; kc < 2; ++kc) {
        v16bf a = load_a_lds(Mw, kc, lane);
#pragma unroll
        for (int nt = 0; nt < 4; ++nt) {
            v16bf b = load_b_lds(ldsWh, H_DIM, kc, nt, lane);
            acc2[nt] = wmma_bf16(a, b, acc2[nt]);
        }
    }
    __syncthreads();
#pragma unroll
    for (int nt = 0; nt < 4; ++nt) {
        const float bias = b_h[nt * 16 + (lane & 15)];
#pragma unroll
        for (int r = 0; r < 8; ++r) {
            float v = fmaxf(acc2[nt][r] + bias, 0.0f);
            const int row = r + 8 * (lane >> 4);
            Mw[row * H_DIM + nt * 16 + (lane & 15)] = (__bf16)v;
        }
    }
    __syncthreads();

    // ---- layer 3: [16,64] @ [64,128] + scatter-add ------------------------
    v8f acc3[8];
#pragma unroll
    for (int nt = 0; nt < 8; ++nt) acc3[nt] = (v8f)0.0f;
#pragma unroll
    for (int kc = 0; kc < 2; ++kc) {
        v16bf a = load_a_lds(Mw, kc, lane);
#pragma unroll
        for (int nt = 0; nt < 8; ++nt) {
            v16bf b = load_b_lds(ldsWout, D_DIM, kc, nt, lane);
            acc3[nt] = wmma_bf16(a, b, acc3[nt]);
        }
    }
#pragma unroll
    for (int nt = 0; nt < 8; ++nt) {
        const float bias = b_out[nt * 16 + (lane & 15)];
#pragma unroll
        for (int r = 0; r < 8; ++r) {
            float v = fmaxf(acc3[nt][r] + bias, 0.0f);
            const int row = r + 8 * (lane >> 4);  // edge within tile
            const int ee = base + row;
            if (ee < E) {
                const int drow = dst[ee];
                unsafeAtomicAdd(&s_out[(size_t)drow * D_DIM + nt * 16 + (lane & 15)], v);
            }
        }
    }
}

// ---------------------------------------------------------------------------
// Molecule aggregation: segment_sum of atom states by mol_id (f32 source).
// ---------------------------------------------------------------------------
__global__ __launch_bounds__(256) void mol_agg_kernel(
    const float* __restrict__ s, const int* __restrict__ mol_ids,
    float* __restrict__ molr, int N)
{
    const int idx = blockIdx.x * blockDim.x + threadIdx.x;
    if (idx >= N * D_DIM) return;
    const int a = idx >> 7;
    const int c = idx & (D_DIM - 1);
    unsafeAtomicAdd(&molr[(size_t)mol_ids[a] * D_DIM + c], s[idx]);
}

// ---------------------------------------------------------------------------
// Head MLP on mol_repr: [256,128]@[128,64] relu @[64,64] relu @[64,32] + b.
// One wave per 16-mol tile; WMMA with on-the-fly f32->bf16 weight fragments.
// ---------------------------------------------------------------------------
__global__ __launch_bounds__(32) void head_mlp_kernel(
    const float* __restrict__ molr,
    const float* __restrict__ fc1w, const float* __restrict__ fc1b,
    const float* __restrict__ fc2w, const float* __restrict__ fc2b,
    const float* __restrict__ outw, const float* __restrict__ outb,
    float* __restrict__ out)
{
    __shared__ __bf16 Mw[16 * H_DIM];
    const int lane = threadIdx.x & 31;
    const int row0 = blockIdx.x * 16;
    const int mrow = lane & 15;
    const int kbase = (lane >> 4) * 8;

    // fc1: K = 128
    v8f acc1[4];
#pragma unroll
    for (int nt = 0; nt < 4; ++nt) acc1[nt] = (v8f)0.0f;
#pragma unroll
    for (int kc = 0; kc < 4; ++kc) {
        const float* r = molr + (size_t)(row0 + mrow) * D_DIM;
        v16bf a;
#pragma unroll
        for (int i = 0; i < 8; ++i) a[i] = (__bf16)r[kc * 32 + kbase + i];
#pragma unroll
        for (int i = 0; i < 8; ++i) a[8 + i] = (__bf16)r[kc * 32 + 16 + kbase + i];
#pragma unroll
        for (int nt = 0; nt < 4; ++nt) {
            v16bf b = load_b_gmem(fc1w, H_DIM, kc, nt, lane);
            acc1[nt] = wmma_bf16(a, b, acc1[nt]);
        }
    }
#pragma unroll
    for (int nt = 0; nt < 4; ++nt) {
        const float bias = fc1b[nt * 16 + (lane & 15)];
#pragma unroll
        for (int r = 0; r < 8; ++r) {
            float v = fmaxf(acc1[nt][r] + bias, 0.0f);
            Mw[(r + 8 * (lane >> 4)) * H_DIM + nt * 16 + (lane & 15)] = (__bf16)v;
        }
    }
    __syncthreads();

    // fc2: K = 64
    v8f acc2[4];
#pragma unroll
    for (int nt = 0; nt < 4; ++nt) acc2[nt] = (v8f)0.0f;
#pragma unroll
    for (int kc = 0; kc < 2; ++kc) {
        v16bf a = load_a_lds(Mw, kc, lane);
#pragma unroll
        for (int nt = 0; nt < 4; ++nt) {
            v16bf b = load_b_gmem(fc2w, H_DIM, kc, nt, lane);
            acc2[nt] = wmma_bf16(a, b, acc2[nt]);
        }
    }
    __syncthreads();
#pragma unroll
    for (int nt = 0; nt < 4; ++nt) {
        const float bias = fc2b[nt * 16 + (lane & 15)];
#pragma unroll
        for (int r = 0; r < 8; ++r) {
            float v = fmaxf(acc2[nt][r] + bias, 0.0f);
            Mw[(r + 8 * (lane >> 4)) * H_DIM + nt * 16 + (lane & 15)] = (__bf16)v;
        }
    }
    __syncthreads();

    // out: K = 64, N = 32
    v8f acc3[2];
#pragma unroll
    for (int nt = 0; nt < 2; ++nt) acc3[nt] = (v8f)0.0f;
#pragma unroll
    for (int kc = 0; kc < 2; ++kc) {
        v16bf a = load_a_lds(Mw, kc, lane);
#pragma unroll
        for (int nt = 0; nt < 2; ++nt) {
            v16bf b = load_b_gmem(outw, 32, kc, nt, lane);
            acc3[nt] = wmma_bf16(a, b, acc3[nt]);
        }
    }
#pragma unroll
    for (int nt = 0; nt < 2; ++nt) {
        const float bias = outb[nt * 16 + (lane & 15)];
#pragma unroll
        for (int r = 0; r < 8; ++r) {
            const int row = row0 + r + 8 * (lane >> 4);
            out[(size_t)row * 32 + nt * 16 + (lane & 15)] = acc3[nt][r] + bias;
        }
    }
}

// ---------------------------------------------------------------------------
extern "C" void kernel_launch(void* const* d_in, const int* in_sizes, int n_in,
                              void* d_out, int out_size, void* d_ws, size_t ws_size,
                              hipStream_t stream) {
    const float* states = (const float*)d_in[0];
    const float* Win    = (const float*)d_in[1];
    const float* b_in   = (const float*)d_in[2];
    const float* Wh     = (const float*)d_in[3];
    const float* b_h    = (const float*)d_in[4];
    const float* Wout   = (const float*)d_in[5];
    const float* b_out  = (const float*)d_in[6];
    const float* fc1w   = (const float*)d_in[7];
    const float* fc1b   = (const float*)d_in[8];
    const float* fc2w   = (const float*)d_in[9];
    const float* fc2b   = (const float*)d_in[10];
    const float* outw   = (const float*)d_in[11];
    const float* outb   = (const float*)d_in[12];
    const int*   src    = (const int*)d_in[13];
    const int*   dst    = (const int*)d_in[14];
    const int*   molid  = (const int*)d_in[15];

    const int N = in_sizes[0] / D_DIM;
    const int E = in_sizes[13];
    const int STEPS = 3;

    // Workspace layout: [ s_f32 : N*128 f32 | s_bf16 : N*128 bf16 | molr ]
    float*  sf32 = (float*)d_ws;
    __bf16* sbf  = (__bf16*)(sf32 + (size_t)N * D_DIM);
    float*  molr = (float*)(sbf + (size_t)N * D_DIM);
    float*  outp = (float*)d_out;

    const int nElem  = N * D_DIM;
    const int n4     = nElem / 4;
    const int cvtBlk = (n4 + 255) / 256;
    const int edgeBlocks = (E + 63) / 64;

    // bf16 mirror of the input states
    cvt_bf16_kernel<<<cvtBlk, 256, 0, stream>>>(states, sbf, n4);

    for (int t = 0; t < STEPS; ++t) {
        hipMemsetAsync(sf32, 0, (size_t)nElem * sizeof(float), stream);
        edge_mlp_kernel<<<edgeBlocks, 128, 0, stream>>>(
            sbf, sf32,
            Win  + (size_t)t * 2 * D_DIM * H_DIM, b_in  + (size_t)t * H_DIM,
            Wh   + (size_t)t * H_DIM * H_DIM,     b_h   + (size_t)t * H_DIM,
            Wout + (size_t)t * H_DIM * D_DIM,     b_out + (size_t)t * D_DIM,
            src, dst, E);
        if (t < STEPS - 1) {
            cvt_bf16_kernel<<<cvtBlk, 256, 0, stream>>>(sf32, sbf, n4);
        }
    }

    hipMemsetAsync(molr, 0, (size_t)256 * D_DIM * sizeof(float), stream);
    mol_agg_kernel<<<(nElem + 255) / 256, 256, 0, stream>>>(sf32, molid, molr, N);

    head_mlp_kernel<<<16, 32, 0, stream>>>(molr, fc1w, fc1b, fc2w, fc2b,
                                           outw, outb, outp);
}
